// RnnModel2_1743756722579
// MI455X (gfx1250) — matrix operand order repaired and checked by reading, not compiled, not run on hardware
//
#include <hip/hip_runtime.h>
#include <hip/hip_bf16.h>
#include <math.h>

// ---------------------------------------------------------------------------
// 3-layer tanh RNN, B=4096, T=512, H=50 (padded to 64), batch-tiled onto
// v_wmma_f32_16x16x32_bf16. Each wave owns 16 batch rows; hidden states are
// 16x64 bf16 tiles in per-wave LDS. Weights pre-swizzled into WMMA B-fragment
// layout in LDS (layer-1's fused matrix hoisted into registers). Bias and the
// scalar input x_t are folded into augmented K-slots 50 / 51 so each layer is
// pure WMMA. A-fragments are carried in registers across layers/iterations.
// All LDS repair stores are divergence-free (duplicate same-value writes).
// ---------------------------------------------------------------------------

typedef __attribute__((ext_vector_type(16))) __bf16 v16bf;
typedef __attribute__((ext_vector_type(8)))  __bf16 v8bf;
typedef __attribute__((ext_vector_type(8)))  float  v8f;

#define HID   50
#define SEQ   512
#define OUTN  10
#define KPAD  64

// weight buffer: 5 matrices x 4 ntiles x 2 kblocks = 40 frags, 512 bf16 each
#define NFRAG 40

__device__ __forceinline__ float fast_tanh(float x) {
#if __has_builtin(__builtin_amdgcn_tanhf)
    return __builtin_amdgcn_tanhf(x);          // v_tanh_f32
#else
    float e = __builtin_amdgcn_exp2f(2.88539008177793f * x);
    return 1.0f - 2.0f * __builtin_amdgcn_rcpf(e + 1.0f);
#endif
}

__device__ __forceinline__ unsigned pack_bf16(float lo, float hi) {
    union { __bf16 h[2]; unsigned u; } p;
    p.h[0] = (__bf16)lo;
    p.h[1] = (__bf16)hi;
    return p.u;
}

// A-fragment (16x32 bf16) from a row-major 16x64 bf16 tile (ISA 7.12.2).
__device__ __forceinline__ v16bf load_afrag(const __bf16* tile, int lane, int kblock) {
    int mrow = lane & 15;
    int hi   = (lane >> 4) & 1;
    const __bf16* base = tile + mrow * KPAD + kblock * 32 + hi * 8;
    v8bf x0 = *(const v8bf*)(base);
    v8bf x1 = *(const v8bf*)(base + 16);
    return __builtin_shufflevector(x0, x1, 0,1,2,3,4,5,6,7,8,9,10,11,12,13,14,15);
}

// B-fragment (32x16 bf16), pre-swizzled: lane L's 16 values are contiguous.
__device__ __forceinline__ v16bf load_bfrag(const __bf16* wbuf, int lane, int f) {
    const __bf16* base = wbuf + f * 512 + lane * 16;
    v8bf x0 = *(const v8bf*)(base);
    v8bf x1 = *(const v8bf*)(base + 8);
    return __builtin_shufflevector(x0, x1, 0,1,2,3,4,5,6,7,8,9,10,11,12,13,14,15);
}

// acc[nt] += A(16x64) x B_m(64x64), B fragments streamed from LDS
__device__ __forceinline__ void gemm_lds(v8f acc[4], const __bf16* wbuf, int lane,
                                         v16bf a0, v16bf a1, int m) {
#pragma unroll
    for (int nt = 0; nt < 4; ++nt) {
        v16bf b0 = load_bfrag(wbuf, lane, m * 8 + nt * 2 + 0);
        v16bf b1 = load_bfrag(wbuf, lane, m * 8 + nt * 2 + 1);
        acc[nt] = __builtin_amdgcn_wmma_f32_16x16x32_bf16(
            false, a0, false, b0, (short)0, acc[nt], false, false);
        acc[nt] = __builtin_amdgcn_wmma_f32_16x16x32_bf16(
            false, a1, false, b1, (short)0, acc[nt], false, false);
    }
}

// acc[nt] += A(16x64) x B(64x64), B fragments resident in registers
__device__ __forceinline__ void gemm_reg(v8f acc[4], const v16bf B[8],
                                         v16bf a0, v16bf a1) {
#pragma unroll
    for (int nt = 0; nt < 4; ++nt) {
        acc[nt] = __builtin_amdgcn_wmma_f32_16x16x32_bf16(
            false, a0, false, B[nt * 2 + 0], (short)0, acc[nt], false, false);
        acc[nt] = __builtin_amdgcn_wmma_f32_16x16x32_bf16(
            false, a1, false, B[nt * 2 + 1], (short)0, acc[nt], false, false);
    }
}

// tanh + store full D tile (C/D layout) into row-major bf16 tile.
// Cols 50..63 receive harmless garbage (their B rows are all zero);
// cols 50/51 are repaired afterwards with a packed b32 store.
__device__ __forceinline__ void store_htile(__bf16* tile, int lane, const v8f acc[4]) {
    int c0 = lane & 15;
    int hi = (lane >> 4) & 1;
#pragma unroll
    for (int nt = 0; nt < 4; ++nt) {
        int col = c0 + nt * 16;
#pragma unroll
        for (int r = 0; r < 8; ++r) {
            int row = r + hi * 8;
            tile[row * KPAD + col] = (__bf16)fast_tanh(acc[nt][r]);
        }
    }
}

__global__ __launch_bounds__(64) void rnn3_wmma_kernel(
    const float* __restrict__ x,
    const float* __restrict__ W_ih1, const float* __restrict__ W_hh1,
    const float* __restrict__ b_ih1, const float* __restrict__ b_hh1,
    const float* __restrict__ W_ih2, const float* __restrict__ W_hh2,
    const float* __restrict__ b_ih2, const float* __restrict__ b_hh2,
    const float* __restrict__ W_ih3, const float* __restrict__ W_hh3,
    const float* __restrict__ b_ih3, const float* __restrict__ b_hh3,
    const float* __restrict__ W_fc,  const float* __restrict__ b_fc,
    float* __restrict__ out)
{
    __shared__ __align__(16) __bf16 wbuf[NFRAG * 512];     // 40 KB
    __shared__ __align__(16) __bf16 htiles[6 * 16 * KPAD]; // 12 KB

    const int tid  = threadIdx.x;
    const int lane = tid & 31;
    const int wave = tid >> 5;
    const int rowbase = blockIdx.x * 32 + wave * 16;

    // ---- prologue 1: build pre-swizzled B-fragments in LDS --------------
    // B element (k, n): out[n] += in[k] * B[k][n].
    // m=0: W_hh1^T, row50 = b_ih1+b_hh1, row51 = W_ih1
    // m=1: W_ih2^T, row50 = b_ih2+b_hh2      m=2: W_hh2^T
    // m=3: W_ih3^T, row50 = b_ih3+b_hh3      m=4: W_hh3^T
    for (int chunk = tid; chunk < NFRAG * 32; chunk += 64) {
        int f = chunk >> 5;
        int s = chunk & 31;              // B-frag lane slot
        int m  = f >> 3;
        int nt = (f >> 1) & 3;
        int kb = f & 1;
        int n  = nt * 16 + (s & 15);
        int kbase = kb * 32 + ((s >> 4) << 3);
#pragma unroll
        for (int j = 0; j < 16; ++j) {
            int k = kbase + (j & 7) + ((j >> 3) << 4);
            float v = 0.0f;
            if (n < HID) {
                switch (m) {
                case 0:
                    if (k < HID)           v = W_hh1[n * HID + k];
                    else if (k == HID)     v = b_ih1[n] + b_hh1[n];
                    else if (k == HID + 1) v = W_ih1[n];   // (H,1) matrix
                    break;
                case 1:
                    if (k < HID)           v = W_ih2[n * HID + k];
                    else if (k == HID)     v = b_ih2[n] + b_hh2[n];
                    break;
                case 2:
                    if (k < HID)           v = W_hh2[n * HID + k];
                    break;
                case 3:
                    if (k < HID)           v = W_ih3[n * HID + k];
                    else if (k == HID)     v = b_ih3[n] + b_hh3[n];
                    break;
                default:
                    if (k < HID)           v = W_hh3[n * HID + k];
                    break;
                }
            }
            wbuf[f * 512 + s * 16 + j] = (__bf16)v;
        }
    }

    // ---- prologue 2: init h tiles (h = 0, col50 = 1.0, col51 = x_0) -----
    for (int idx = tid; idx < 6 * 16 * KPAD; idx += 64) {
        int tile = idx >> 10;            // wave*3 + layer
        int e    = idx & 1023;
        int row  = e >> 6;
        int col  = e & 63;
        int w    = tile / 3;
        int l    = tile % 3;
        float v = 0.0f;
        if (col == HID) v = 1.0f;
        else if (col == HID + 1 && l == 0)
            v = x[(size_t)(blockIdx.x * 32 + w * 16 + row) * SEQ + 0];
        htiles[idx] = (__bf16)v;
    }
    __syncthreads();

    __bf16* h1 = &htiles[(wave * 3 + 0) * 16 * KPAD];
    __bf16* h2 = &htiles[(wave * 3 + 1) * 16 * KPAD];
    __bf16* h3 = &htiles[(wave * 3 + 2) * 16 * KPAD];

    // hoist layer-1 fused B matrix (8 fragments = 64 VGPRs) into registers
    v16bf B1[8];
#pragma unroll
    for (int f = 0; f < 8; ++f) B1[f] = load_bfrag(wbuf, lane, f);

    // carried A-fragments (reloaded once per step, right after each store)
    v16bf A1k0 = load_afrag(h1, lane, 0), A1k1 = load_afrag(h1, lane, 1);
    v16bf A2k0 = load_afrag(h2, lane, 0), A2k1 = load_afrag(h2, lane, 1);
    v16bf A3k0 = load_afrag(h3, lane, 0), A3k1 = load_afrag(h3, lane, 1);

    const unsigned one_zero = pack_bf16(1.0f, 0.0f);
    const int xrow = lane & 15;          // all 32 lanes mirror rows 0..15

    // ---- recurrence: 512 steps, 40 WMMAs per step per wave --------------
    for (int t = 0; t < SEQ; ++t) {
        // x_{t+1} for this wave's 16 rows (lanes L and L+16 duplicate)
        int tn = (t + 1 < SEQ) ? (t + 1) : (SEQ - 1);
        float xn = x[(size_t)(rowbase + xrow) * SEQ + tn];

        // layer 1: h1' = tanh([h1 | 1 | x_t] @ B1)   (B1 in registers)
        {
            v8f acc[4] = {};
            gemm_reg(acc, B1, A1k0, A1k1);
            store_htile(h1, lane, acc);
        }
        *(unsigned*)(h1 + xrow * KPAD + HID) = pack_bf16(1.0f, xn);
        A1k0 = load_afrag(h1, lane, 0);
        A1k1 = load_afrag(h1, lane, 1);

        // layer 2: h2' = tanh([h1'|1] @ B_ih2 + h2 @ B_hh2)
        {
            v8f acc[4] = {};
            gemm_lds(acc, wbuf, lane, A1k0, A1k1, 1);
            gemm_lds(acc, wbuf, lane, A2k0, A2k1, 2);
            store_htile(h2, lane, acc);
        }
        *(unsigned*)(h2 + xrow * KPAD + HID) = one_zero;
        A2k0 = load_afrag(h2, lane, 0);
        A2k1 = load_afrag(h2, lane, 1);

        // layer 3: h3' = tanh([h2'|1] @ B_ih3 + h3 @ B_hh3)
        {
            v8f acc[4] = {};
            gemm_lds(acc, wbuf, lane, A2k0, A2k1, 3);
            gemm_lds(acc, wbuf, lane, A3k0, A3k1, 4);
            store_htile(h3, lane, acc);
        }
        *(unsigned*)(h3 + xrow * KPAD + HID) = one_zero;
        A3k0 = load_afrag(h3, lane, 0);
        A3k1 = load_afrag(h3, lane, 1);
    }

    // ---- final FC: out[row] = h3 @ W_fc^T + b_fc (10 outputs) -----------
    // lane L: row = L%16, outputs [5*(L/16), 5*(L/16)+4]
    {
        int row = lane & 15;
        int ob  = (lane >> 4) * 5;
        int gr  = rowbase + row;
#pragma unroll
        for (int oo = 0; oo < 5; ++oo) {
            int o = ob + oo;
            float s = b_fc[o];
            for (int k = 0; k < HID; ++k)
                s += (float)h3[row * KPAD + k] * W_fc[o * HID + k];
            out[(size_t)gr * OUTN + o] = s;
        }
    }
}

extern "C" void kernel_launch(void* const* d_in, const int* in_sizes, int n_in,
                              void* d_out, int out_size, void* d_ws, size_t ws_size,
                              hipStream_t stream) {
    (void)in_sizes; (void)n_in; (void)d_ws; (void)ws_size; (void)out_size;
    const float* x     = (const float*)d_in[0];
    const float* W_ih1 = (const float*)d_in[1];
    const float* W_hh1 = (const float*)d_in[2];
    const float* b_ih1 = (const float*)d_in[3];
    const float* b_hh1 = (const float*)d_in[4];
    const float* W_ih2 = (const float*)d_in[5];
    const float* W_hh2 = (const float*)d_in[6];
    const float* b_ih2 = (const float*)d_in[7];
    const float* b_hh2 = (const float*)d_in[8];
    const float* W_ih3 = (const float*)d_in[9];
    const float* W_hh3 = (const float*)d_in[10];
    const float* b_ih3 = (const float*)d_in[11];
    const float* b_hh3 = (const float*)d_in[12];
    const float* W_fc  = (const float*)d_in[13];
    const float* b_fc  = (const float*)d_in[14];

    // 4096 rows / (2 waves * 16 rows) = 128 workgroups of 64 threads
    rnn3_wmma_kernel<<<128, 64, 0, stream>>>(
        x, W_ih1, W_hh1, b_ih1, b_hh1,
        W_ih2, W_hh2, b_ih2, b_hh2,
        W_ih3, W_hh3, b_ih3, b_hh3,
        W_fc, b_fc, (float*)d_out);
}
